// ResidualVectorQuantize_57913339020075
// MI455X (gfx1250) — matrix operand order repaired and checked by reading, not compiled, not run on hardware
//
#include <hip/hip_runtime.h>
#include <hip/hip_bf16.h>

// ---------------------------------------------------------------------------
// Residual Vector Quantize, MI455X / gfx1250 (wave32, WMMA).
// Pipeline: prep (normalize codebooks, transpose weights) -> rearrange
// (pre_process + f32->bf16) -> down GEMM (bf16 WMMA) -> fused 6-stream RVQ
// (bf16 WMMA search + argmax, residual in LDS, async-to-LDS codebook staging)
// -> up GEMM (f32 WMMA) fused with post_process scatter -> deterministic
// loss reduction.
// ---------------------------------------------------------------------------

#define DEVFN __device__ __forceinline__

typedef __attribute__((ext_vector_type(16))) __bf16 v16bf;
typedef __attribute__((ext_vector_type(8)))  float  v8f;
typedef __attribute__((ext_vector_type(2)))  float  v2f;

// GCC-style int4 vector: the async-LDS builtin's pointee type
// (diagnostic showed: '__attribute__((__vector_size__(4*sizeof(int)))) int __device__ *')
typedef int gv4i __attribute__((vector_size(16)));
typedef __attribute__((address_space(1))) gv4i gv4i_g;   // global (AS1)
typedef __attribute__((address_space(3))) gv4i gv4i_l;   // LDS (AS3)

#if __has_builtin(__builtin_amdgcn_global_load_async_to_lds_b128)
#define HAVE_ASYNC_LDS 1
#else
#define HAVE_ASYNC_LDS 0
#endif

DEVFN void wait_asynccnt0() {
#if __has_builtin(__builtin_amdgcn_s_wait_asynccnt)
    __builtin_amdgcn_s_wait_asynccnt(0);
#else
    asm volatile("s_wait_asynccnt 0" ::: "memory");
#endif
}

// Problem constants
constexpr int B_      = 32;
constexpr int C_      = 64;
constexpr int H_      = 6;
constexpr int W_      = 4096;
constexpr int FIX     = C_ * H_;        // 384
constexpr int HIDDEN  = FIX * 4;        // 1536
constexpr int T_      = W_ / 4;         // 1024
constexpr int ROWS    = B_ * T_;        // 32768
constexpr int CB_DIM  = 64;
constexpr int CB_SIZE = 1024;

// Output layout (floats)
constexpr long OUT_Z     = 0;                         // 32*24576*64
constexpr long OUT_CODES = (long)B_ * H_ * W_ * C_;   // 50331648
constexpr long OUT_LOSS  = OUT_CODES + (long)B_ * 6 * T_; // 50528256

// Workspace layout (bytes)
constexpr size_t OFF_XPRE = 0;                                   // bf16 32768x1536
constexpr size_t OFF_X    = OFF_XPRE + (size_t)ROWS * HIDDEN * 2; // f32 32768x64
constexpr size_t OFF_ZQ   = OFF_X    + (size_t)ROWS * CB_DIM * 4;
constexpr size_t OFF_WDT  = OFF_ZQ   + (size_t)ROWS * CB_DIM * 4; // bf16 64x1536 (N-major)
constexpr size_t OFF_WUT  = OFF_WDT  + (size_t)CB_DIM * HIDDEN * 2; // f32 1536x64 (N-major)
constexpr size_t OFF_ENN  = OFF_WUT  + (size_t)HIDDEN * CB_DIM * 4; // f32 6x1024x64 normalized
constexpr size_t OFF_ENB  = OFF_ENN  + (size_t)6 * CB_SIZE * CB_DIM * 4; // bf16 same
constexpr size_t OFF_LP   = OFF_ENB  + (size_t)6 * CB_SIZE * CB_DIM * 2; // f32[256] partials

// ---------------------------------------------------------------------------
// helpers
// ---------------------------------------------------------------------------
DEVFN unsigned short f2bf(float x) {
    unsigned u = __builtin_bit_cast(unsigned, x);
    unsigned r = u + 0x7FFFu + ((u >> 16) & 1u);   // round-to-nearest-even
    return (unsigned short)(r >> 16);
}

struct U128x2 { uint4 lo, hi; };
struct U16x16 { unsigned short u[16]; };

// Load a 32-byte WMMA fragment as two 16B chunks (global or LDS pointers).
DEVFN v16bf ld_frag(const unsigned short* p0, const unsigned short* p1) {
    U128x2 u;
    u.lo = *(const uint4*)p0;
    u.hi = *(const uint4*)p1;
    return __builtin_bit_cast(v16bf, u);
}

// A fragment built from an fp32 row (LDS residual): element e -> K=kb+koff+e,
// element 8+e -> K=kb+16+koff+e  (koff = lane>=16 ? 8 : 0).
DEVFN v16bf cvt_a_frag(const float* row, int kb, int koff) {
    U16x16 t;
#pragma unroll
    for (int e = 0; e < 8; ++e) {
        t.u[e]     = f2bf(row[kb + koff + e]);
        t.u[8 + e] = f2bf(row[kb + 16 + koff + e]);
    }
    return __builtin_bit_cast(v16bf, t);
}

DEVFN v8f wmma_bf16(v16bf a, v16bf b, v8f c) {
    return __builtin_amdgcn_wmma_f32_16x16x32_bf16(false, a, false, b,
                                                   (short)0, c, false, false);
}
DEVFN v8f wmma_f32(v2f a, v2f b, v8f c) {
    return __builtin_amdgcn_wmma_f32_16x16x4_f32(false, a, false, b,
                                                 (short)0, c, false, false);
}

// ---------------------------------------------------------------------------
// Kernel 0: prep — normalize codebooks (f32 + bf16 copies), transpose W_down
// to bf16 N-major, transpose W_up to f32 N-major.
// ---------------------------------------------------------------------------
__global__ void prep_kernel(const float* __restrict__ Wd, const float* __restrict__ Wu,
                            const float* __restrict__ emb,
                            unsigned short* __restrict__ wdT, float* __restrict__ wuT,
                            float* __restrict__ enN, unsigned short* __restrict__ enB) {
    const int tid = threadIdx.x;
    const int blk = blockIdx.x;
    if (blk < 768) {                               // 6144 codebook rows, 1 wave each
        const int wave = blk * 8 + (tid >> 5);
        const int lane = tid & 31;
        const float* row = emb + (long)wave * 64;
        float e0 = row[lane], e1 = row[lane + 32];
        float ss = e0 * e0 + e1 * e1;
#pragma unroll
        for (int off = 16; off; off >>= 1) ss += __shfl_xor(ss, off, 32);
        float scale = 1.0f / fmaxf(sqrtf(ss), 1e-12f);
        float n0 = e0 * scale, n1 = e1 * scale;
        enN[(long)wave * 64 + lane]      = n0;
        enN[(long)wave * 64 + lane + 32] = n1;
        enB[(long)wave * 64 + lane]      = f2bf(n0);
        enB[(long)wave * 64 + lane + 32] = f2bf(n1);
    } else if (blk < 1152) {                       // W_down^T bf16: wdT[n][k]
        const int i = (blk - 768) * 256 + tid;     // < 98304
        const int n = i / HIDDEN, k = i % HIDDEN;
        wdT[(long)n * HIDDEN + k] = f2bf(Wd[(long)k * CB_DIM + n]);
    } else {                                       // W_up^T f32: wuT[n][k]
        const int i = (blk - 1152) * 256 + tid;    // < 98304
        const int n = i >> 6, k = i & 63;
        wuT[i] = Wu[(long)k * HIDDEN + n];
    }
}

// ---------------------------------------------------------------------------
// Kernel 1: pre_process rearrange, f32 -> bf16.
// x_pre[b*T+t][o*384 + c*6 + h] = z_e[(b*24576 + h*4096 + w)*64 + c], w=4t+o
// ---------------------------------------------------------------------------
__global__ void rearrange_kernel(const float* __restrict__ ze,
                                 unsigned short* __restrict__ xpre) {
    const long idx = (long)blockIdx.x * 256 + threadIdx.x;   // < 8388608 = B*W*C
    const int c = (int)(idx & 63);
    const int w = (int)((idx >> 6) & 4095);
    const int b = (int)(idx >> 18);
    const int t = w >> 2, o = w & 3;
    unsigned short* dst = xpre + ((long)(b * T_ + t) * HIDDEN + o * FIX + c * H_);
    const float* src = ze + ((long)b * (H_ * W_) + w) * C_ + c;
#pragma unroll
    for (int h = 0; h < H_; ++h)
        dst[h] = f2bf(src[(long)h * W_ * C_]);
}

// ---------------------------------------------------------------------------
// Kernel 2: down GEMM  x = x_pre(bf16) @ W_down  -> f32 (32768 x 64)
// One wave per 16x16 output tile; K = 1536 in steps of 32 (bf16 WMMA).
// ---------------------------------------------------------------------------
__global__ void down_gemm_kernel(const unsigned short* __restrict__ xpre,
                                 const unsigned short* __restrict__ wdT,
                                 float* __restrict__ x) {
    const int tid = threadIdx.x, lane = tid & 31, wv = tid >> 5;
    const int task = blockIdx.x * 8 + wv;          // 8192 tiles
    const int rowBase = (task >> 2) * 16;
    const int colBase = (task & 3) * 16;
    const int l16  = lane & 15;
    const int hi   = lane >> 4;
    const int koff = hi ? 8 : 0;                   // A pattern
    const int boff = hi ? 16 : 0;                  // unused: B loads use koff
    const unsigned short* arow = xpre + (long)(rowBase + l16) * HIDDEN;
    const unsigned short* brow = wdT  + (long)(colBase + l16) * HIDDEN;
    v8f acc = {};
    for (int kb = 0; kb < HIDDEN; kb += 32) {
        __builtin_prefetch(arow + kb + 256, 0, 3);   // global_prefetch a cacheline ahead
        v16bf a = ld_frag(arow + kb + koff, arow + kb + 16 + koff);
        v16bf b = ld_frag(brow + kb + koff, brow + kb + 16 + koff);
        acc = wmma_bf16(a, b, acc);
    }
    const int mofs = hi ? 8 : 0;
#pragma unroll
    for (int v = 0; v < 8; ++v)
        x[(long)(rowBase + v + mofs) * CB_DIM + colBase + l16] = acc[v];
}

// ---------------------------------------------------------------------------
// Kernel 3: fused residual VQ (ns streams). 1 wave = 16 rows, residual in LDS.
// Search: scores = residual @ en^T via bf16 WMMA (argmax is norm-invariant),
// codebook staged through a 16KB LDS window in 8 chunks of 128 codes using
// the CDNA5 async global->LDS path (ASYNCcnt) when available.
// ---------------------------------------------------------------------------
__global__ void rvq_kernel(const float* __restrict__ xin,
                           const float* __restrict__ enN,
                           const unsigned short* __restrict__ enB,
                           const int* __restrict__ ns_ptr,
                           float* __restrict__ zq_out,
                           float* __restrict__ codes_out,
                           float* __restrict__ loss_partial) {
    __shared__ __align__(16) unsigned short en_s[128 * 64];   // 16 KB chunk
    __shared__ __align__(16) float res_s[8 * 16 * 64];        // 32 KB residuals
    __shared__ int   code_s[8 * 16];
    __shared__ float loss_s[8];

    const int tid  = threadIdx.x;
    const int lane = tid & 31;
    const int wv   = tid >> 5;
    const int ns   = ns_ptr[0];
    const int rowBase = (blockIdx.x * 8 + wv) * 16;
    float* myres = res_s + wv * 16 * 64;

    {   // initial residual = x tile (16 x 64 f32), vectorized copy
        const float4* src = (const float4*)(xin + (long)rowBase * CB_DIM);
        float4* dst = (float4*)myres;
        for (int j = lane; j < 256; j += 32) dst[j] = src[j];
    }

    const int l16  = lane & 15;
    const int hi   = lane >> 4;
    const int koff = hi ? 8 : 0;    // A fragment K offset
    const int boff = hi ? 16 : 0;   // unused: B fragments use koff

    float zq[32];
#pragma unroll
    for (int j = 0; j < 32; ++j) zq[j] = 0.0f;
    float lossAcc = 0.0f;

    for (int s = 0; s < ns; ++s) {
        // A fragments for this wave's 16 rows (K = 0..63 as two k-steps)
        v16bf a0 = cvt_a_frag(myres + l16 * 64, 0,  koff);
        v16bf a1 = cvt_a_frag(myres + l16 * 64, 32, koff);

        float best[8]; int bidx[8];
#pragma unroll
        for (int v = 0; v < 8; ++v) { best[v] = -3.4e38f; bidx[v] = 0; }

        for (int ch = 0; ch < 8; ++ch) {           // 8 x 128 codebook rows
            __syncthreads();                       // protect en_s reuse
            {   // cooperative stage: 128 rows * 64 bf16 = 1024 x b128
                const unsigned short* gsrc = enB + ((long)s * CB_SIZE + ch * 128) * 64;
#if HAVE_ASYNC_LDS
                // CDNA5 async global->LDS DMA: no VGPR round-trip, ASYNCcnt.
                for (int j = tid; j < 1024; j += 256) {
                    __builtin_amdgcn_global_load_async_to_lds_b128(
                        (gv4i_g*)(gsrc + j * 8),
                        (gv4i_l*)(en_s + j * 8),
                        0, 0);
                }
                wait_asynccnt0();
#else
                const uint4* gs4 = (const uint4*)gsrc;
                uint4* gdst = (uint4*)en_s;
                for (int j = tid; j < 1024; j += 256) gdst[j] = gs4[j];
#endif
            }
            __syncthreads();
#pragma unroll
            for (int nt = 0; nt < 8; ++nt) {
                const unsigned short* brow = en_s + (nt * 16 + l16) * 64;
                v16bf b0 = ld_frag(brow + koff,      brow + 16 + koff);
                v16bf b1 = ld_frag(brow + 32 + koff, brow + 32 + 16 + koff);
                v8f acc = {};
                acc = wmma_bf16(a0, b0, acc);
                acc = wmma_bf16(a1, b1, acc);
                const int cidx = ch * 128 + nt * 16 + l16;
#pragma unroll
                for (int v = 0; v < 8; ++v) {
                    float sc = acc[v];
                    if (sc > best[v]) { best[v] = sc; bidx[v] = cidx; }
                }
            }
        }
        // per-row argmax across the 16 columns held in each half-wave;
        // tie-break to the lowest index (jnp.argmax semantics)
#pragma unroll
        for (int off = 8; off; off >>= 1) {
#pragma unroll
            for (int v = 0; v < 8; ++v) {
                float ov = __shfl_xor(best[v], off, 16);
                int   oi = __shfl_xor(bidx[v], off, 16);
                if (ov > best[v] || (ov == best[v] && oi < bidx[v])) {
                    best[v] = ov; bidx[v] = oi;
                }
            }
        }
        if (l16 == 0) {                            // lanes 0 & 16 publish codes
            const int mofs = hi ? 8 : 0;
#pragma unroll
            for (int v = 0; v < 8; ++v) {
                const int m  = v + mofs;
                code_s[wv * 16 + m] = bidx[v];
                const int rg = rowBase + m;
                const int bb = rg >> 10, tt = rg & 1023;
                codes_out[(bb * ns + s) * T_ + tt] = (float)bidx[v];
            }
        }
        __syncthreads();
        // residual update + loss: lane handles row l16, dims [hi*32, hi*32+32)
        const int code = code_s[wv * 16 + l16];
        const float* cvec = enN + ((long)(s << 10) + code) * 64 + (hi << 5);
        float* rptr = myres + l16 * 64 + (hi << 5);
#pragma unroll
        for (int j = 0; j < 32; ++j) {
            float cv = cvec[j];
            float ro = rptr[j];
            float d  = cv - ro;          // (z_q_i - residual): cb == cm forward
            lossAcc += d * d;
            rptr[j]  = ro - cv;
            zq[j]   += cv;
        }
        __syncthreads();
    }

    {   // write accumulated z_q
        float* dst = zq_out + (long)(rowBase + l16) * CB_DIM + (hi << 5);
#pragma unroll
        for (int j = 0; j < 32; ++j) dst[j] = zq[j];
    }
    // deterministic loss reduction: wave -> block partial
#pragma unroll
    for (int off = 16; off; off >>= 1) lossAcc += __shfl_xor(lossAcc, off, 32);
    if (lane == 0) loss_s[wv] = lossAcc;
    __syncthreads();
    if (tid == 0) {
        float t = 0.f;
#pragma unroll
        for (int i = 0; i < 8; ++i) t += loss_s[i];
        loss_partial[blockIdx.x] = t;
    }
}

// ---------------------------------------------------------------------------
// Kernel 4: up GEMM  out = z_q @ W_up (f32 WMMA 16x16x4, K=64) fused with
// post_process scatter. 1 block = 1 row-tile, 8 waves x 12 col-tiles.
// ---------------------------------------------------------------------------
__global__ void up_gemm_kernel(const float* __restrict__ zq,
                               const float* __restrict__ wuT,
                               float* __restrict__ out) {
    const int tid = threadIdx.x, lane = tid & 31, wv = tid >> 5;
    const int rowBase = blockIdx.x * 16;
    const int l16  = lane & 15;
    const int hi   = lane >> 4;
    const int koff = hi ? 2 : 0;
    const float* arow = zq + (long)(rowBase + l16) * CB_DIM;
    v2f a[16];
#pragma unroll
    for (int s2 = 0; s2 < 16; ++s2) {
        v2f t = { arow[s2 * 4 + koff], arow[s2 * 4 + koff + 1] };
        a[s2] = t;
    }
    const int mofs = hi ? 8 : 0;
    for (int ct = wv * 12; ct < wv * 12 + 12; ++ct) {
        const int N = ct * 16 + l16;
        const float* brow = wuT + (long)N * CB_DIM;
        v8f acc = {};
#pragma unroll
        for (int s2 = 0; s2 < 16; ++s2) {
            v2f b = { brow[s2 * 4 + koff], brow[s2 * 4 + koff + 1] };
            acc = wmma_f32(a[s2], b, acc);
        }
        // post_process: N -> (o, c, h);  out[(b*24576 + h*4096 + w)*64 + c]
        const int o = N / FIX, rem = N % FIX, cc = rem / H_, hh = rem % H_;
#pragma unroll
        for (int v = 0; v < 8; ++v) {
            const int rg = rowBase + v + mofs;
            const int bb = rg >> 10, tt = rg & 1023;
            const int ww = 4 * tt + o;
            out[((long)bb * (H_ * W_) + (long)hh * W_ + ww) * C_ + cc] = acc[v];
        }
    }
}

// ---------------------------------------------------------------------------
// Kernel 5: final deterministic loss sum; cb_total == cm_total in forward.
// ---------------------------------------------------------------------------
__global__ void loss_final_kernel(const float* __restrict__ partials,
                                  float* __restrict__ out_losses) {
    __shared__ float s[256];
    const int tid = threadIdx.x;
    s[tid] = partials[tid];
    __syncthreads();
    for (int off = 128; off; off >>= 1) {
        if (tid < off) s[tid] += s[tid + off];
        __syncthreads();
    }
    if (tid == 0) {
        const float m = s[0] / (float)((long)ROWS * CB_DIM);  // sum of per-stream means
        out_losses[0] = m;
        out_losses[1] = m;
    }
}

// ---------------------------------------------------------------------------
extern "C" void kernel_launch(void* const* d_in, const int* in_sizes, int n_in,
                              void* d_out, int out_size, void* d_ws, size_t ws_size,
                              hipStream_t stream) {
    (void)in_sizes; (void)n_in; (void)out_size; (void)ws_size;
    const float* z_e    = (const float*)d_in[0];
    const float* W_down = (const float*)d_in[1];
    const float* W_up   = (const float*)d_in[2];
    const float* emb    = (const float*)d_in[3];
    const int*   ns_p   = (const int*)d_in[4];

    unsigned char* ws = (unsigned char*)d_ws;
    unsigned short* xpre = (unsigned short*)(ws + OFF_XPRE);
    float*          x    = (float*)(ws + OFF_X);
    float*          zq   = (float*)(ws + OFF_ZQ);
    unsigned short* wdT  = (unsigned short*)(ws + OFF_WDT);
    float*          wuT  = (float*)(ws + OFF_WUT);
    float*          enN  = (float*)(ws + OFF_ENN);
    unsigned short* enB  = (unsigned short*)(ws + OFF_ENB);
    float*          lossP= (float*)(ws + OFF_LP);

    float* out       = (float*)d_out;
    float* codes_out = out + OUT_CODES;
    float* loss_out  = out + OUT_LOSS;

    prep_kernel      <<<1536,  256, 0, stream>>>(W_down, W_up, emb, wdT, wuT, enN, enB);
    rearrange_kernel <<<32768, 256, 0, stream>>>(z_e, xpre);
    down_gemm_kernel <<<1024,  256, 0, stream>>>(xpre, wdT, x);
    rvq_kernel       <<<256,   256, 0, stream>>>(x, enN, enB, ns_p, zq, codes_out, lossP);
    up_gemm_kernel   <<<2048,  256, 0, stream>>>(zq, wuT, out);
    loss_final_kernel<<<1,     256, 0, stream>>>(lossP, loss_out);
}
